// NetVLADLayer_48490180772366
// MI455X (gfx1250) — matrix operand order repaired and checked by reading, not compiled, not run on hardware
//
#include <hip/hip_runtime.h>
#include <hip/hip_bf16.h>
#include <math.h>

// ---------------- problem constants ----------------
#define B_ 16
#define D_ 512
#define K_ 64
#define N_ 4096
#define NCHUNK 16
#define NC (N_ / NCHUNK)   // 256 columns per workgroup
#define NT 32              // columns per subtile (== WMMA K for the agg GEMM)
#define NSUB (NC / NT)     // 8 subtiles

typedef __attribute__((ext_vector_type(16))) __bf16        v16bf;
typedef __attribute__((ext_vector_type(8)))  float         v8f;
typedef __attribute__((ext_vector_type(4)))  unsigned int  v4u;
typedef __attribute__((ext_vector_type(8)))  int           v8i;
typedef __attribute__((ext_vector_type(4)))  int           v4i;

union Frag {
  v16bf  v;
  float4 q[2];   // two 16-byte halves (elements 0..7 / 8..15)
};

__device__ __forceinline__ __bf16 f2bf(float f) {
  union { float f; unsigned u; } a; a.f = f;
  unsigned r = a.u + 0x7FFFu + ((a.u >> 16) & 1u);   // round-to-nearest-even
  union { unsigned short s; __bf16 b; } o; o.s = (unsigned short)(r >> 16);
  return o.b;
}

// ---------------------------------------------------------------------------
// TDM: async DMA of a 2D f32 tile [NT cols x D_ rows] (row stride N_) into LDS.
// D# packing per CDNA5 ISA §8.3/8.4 (group0: count/lds/global/type,
// group1: data_size, tensor dims, tile dims, dim0 stride). 2D => groups 2/3 zero.
// This toolchain uses the 6-arg builtin: (g0, g1, g2, g3, g_ext, cpol).
// ---------------------------------------------------------------------------
__device__ __forceinline__ void tdm_load_tile(const float* gptr, unsigned lds_off) {
  unsigned long long ga = (unsigned long long)(const void*)gptr;
  v4u g0;
  g0.x = 1u;                                                   // count=1, user D#
  g0.y = lds_off;                                              // lds_addr (bytes)
  g0.z = (unsigned)ga;                                         // global_addr[31:0]
  g0.w = (unsigned)((ga >> 32) & 0x01FFFFFFu) | (2u << 30);    // addr[56:32] | type=2
  v8i g1;
  g1[0] = (2 << 16);                                           // data_size = 4B; no mcast
  g1[1] = (int)((unsigned)(N_ & 0xFFFF) << 16);                // tensor_dim0[15:0]
  g1[2] = (int)(((N_ >> 16) & 0xFFFF) | ((D_ & 0xFFFF) << 16));// dim0[31:16] | dim1[15:0]
  g1[3] = (int)(((D_ >> 16) & 0xFFFF) | (NT << 16));           // dim1[31:16] | tile_dim0
  g1[4] = D_;                                                  // tile_dim1 | tile_dim2=0
  g1[5] = N_;                                                  // tensor_dim0_stride[31:0]
  g1[6] = 0;                                                   // stride hi | dim1_stride lo
  g1[7] = 0;
  v4i z4 = (v4i){0, 0, 0, 0};
  v8i z8 = (v8i){0, 0, 0, 0, 0, 0, 0, 0};
  __builtin_amdgcn_tensor_load_to_lds(g0, g1, z4, z4, z8, 0);
}

// ---------------------------------------------------------------------------
// Kernel 0: zero the f32 accumulation workspace (agg, mass, gss)
// ---------------------------------------------------------------------------
__global__ void netvlad_zero(float* __restrict__ p, int n) {
  int i = blockIdx.x * blockDim.x + threadIdx.x;
  if (i < n) p[i] = 0.0f;
}

// ---------------------------------------------------------------------------
// Kernel 1: fused  logits GEMM -> softmax -> aggregation GEMM  (one pass on x)
//   grid  = (NCHUNK, B), block = 256 threads (8 wave32)
//   dynamic LDS (268 KB):
//     Wbf[64][512] bf16 | stage[2][512][32] f32 (TDM double buffer) |
//     xdn[512][32] bf16 | xnd[32][512] bf16 | scoresF[64][32] f32 | sbf bf16
// ---------------------------------------------------------------------------
#define OFF_STAGE   65536
#define OFF_XDN    196608
#define OFF_XND    229376
#define OFF_SCORES 262144
#define OFF_SBF    270336
#define SH_BYTES   274432

__global__ __launch_bounds__(256, 1)
void netvlad_fused(const float* __restrict__ x,     // [B,D,N]
                   const float* __restrict__ W,     // [K,D]
                   float* __restrict__ agg,         // [B,D,K] (atomic accum)
                   float* __restrict__ mass)        // [B,K]   (atomic accum)
{
  extern __shared__ char smem[];
  __bf16* Wbf     = (__bf16*)(smem);
  float*  stage   = (float *)(smem + OFF_STAGE);    // 2 x [512][32] f32
  __bf16* xdn     = (__bf16*)(smem + OFF_XDN);      // [d][n]
  __bf16* xnd     = (__bf16*)(smem + OFF_XND);      // [n][d]
  float*  scoresF = (float *)(smem + OFF_SCORES);   // [k][n] f32
  __bf16* sbf     = (__bf16*)(smem + OFF_SBF);      // [k][n] bf16

  const int tid  = threadIdx.x;
  const int wave = tid >> 5;
  const int lane = tid & 31;
  const int hi   = lane >> 4;      // which 16-lane half
  const int lr   = lane & 15;
  const int b    = blockIdx.y;
  const int n0   = blockIdx.x * NC;

  // kick off TDM prefetch of subtile 0 before anything else (wave 0 only)
  if (wave == 0) {
    tdm_load_tile(x + (size_t)b * D_ * N_ + n0, OFF_STAGE);
  }

  // stage W as bf16 (128 KB read, resident in L2 across all 256 WGs)
  for (int idx = tid; idx < K_ * D_; idx += 256) Wbf[idx] = f2bf(W[idx]);

  // per-wave agg accumulators: D rows [wave*64 .. wave*64+63] x all K=64
  v8f acc[4][4];
  #pragma unroll
  for (int i = 0; i < 4; ++i)
    #pragma unroll
    for (int j = 0; j < 4; ++j)
      acc[i][j] = (v8f){0.f,0.f,0.f,0.f,0.f,0.f,0.f,0.f};

  float massReg = 0.0f;            // threads 0..63 own mass[k=tid]
  const int kt = wave & 3;         // logits output tile row (K dim)
  const int nt = wave >> 2;        // logits output tile col (n dim)

  for (int sub = 0; sub < NSUB; ++sub) {
    // ---- wave 0: issue next TDM load, then wait for current buffer ----
    if (wave == 0) {
      if (sub + 1 < NSUB) {
        unsigned dst = OFF_STAGE + ((unsigned)((sub + 1) & 1)) * (D_ * NT * 4);
        tdm_load_tile(x + (size_t)b * D_ * N_ + n0 + (sub + 1) * NT, dst);
        __builtin_amdgcn_s_wait_tensorcnt(1);   // current tile complete
      } else {
        __builtin_amdgcn_s_wait_tensorcnt(0);
      }
    }
    __syncthreads();   // stage[cur] ready for all; xdn/xnd free from prev iter

    // ---- convert staged f32 tile -> bf16 [d][n] and [n][d] copies ----
    const float* stg = stage + (sub & 1) * (D_ * NT);
    for (int idx = tid; idx < D_ * NT; idx += 256) {
      int d = idx >> 5, j = idx & 31;
      __bf16 xb = f2bf(stg[idx]);          // stage is [d][32] row-major
      xdn[d * NT + j] = xb;
      xnd[j * D_ + d] = xb;
    }
    __syncthreads();

    // ---- logits tile: wave computes 16x16 tile (kt,nt) over D=512 ----
    v8f lc = (v8f){0.f,0.f,0.f,0.f,0.f,0.f,0.f,0.f};
    #pragma unroll
    for (int ks = 0; ks < D_ / 32; ++ks) {
      Frag fa, fb;
      // A = W[16 x 32]: lane half picks K-subsets {0..7,16..23} / {8..15,24..31}
      const __bf16* ar = Wbf + (kt * 16 + lr) * D_ + ks * 32;
      fa.q[0] = *(const float4*)(ar + 8 * hi);
      fa.q[1] = *(const float4*)(ar + 16 + 8 * hi);
      // B = x[32 x 16] from [n][d] copy: lane half picks K(d)-subsets {0..7,16..23} / {8..15,24..31}
      const __bf16* br = xnd + (nt * 16 + lr) * D_ + ks * 32;
      fb.q[0] = *(const float4*)(br + 8 * hi);
      fb.q[1] = *(const float4*)(br + 16 + 8 * hi);
      lc = __builtin_amdgcn_wmma_f32_16x16x32_bf16(false, fa.v, false, fb.v,
                                                   (short)0, lc, false, false);
    }
    // C/D layout: VGPR e -> M = e + 8*hi, N = lane&15
    #pragma unroll
    for (int e = 0; e < 8; ++e)
      scoresF[(kt * 16 + e + 8 * hi) * NT + nt * 16 + lr] = lc[e];
    __syncthreads();

    // ---- softmax over K (64) per column, f32 in LDS ----
    if (tid < NT) {
      float m = -3.0e38f;
      for (int k = 0; k < K_; ++k) m = fmaxf(m, scoresF[k * NT + tid]);
      float s = 0.0f;
      for (int k = 0; k < K_; ++k) {
        float e = __expf(scoresF[k * NT + tid] - m);
        scoresF[k * NT + tid] = e; s += e;
      }
      float r = 1.0f / s;
      for (int k = 0; k < K_; ++k) {
        float p = scoresF[k * NT + tid] * r;
        scoresF[k * NT + tid] = p;
        sbf[k * NT + tid] = f2bf(p);
      }
    }
    __syncthreads();

    // ---- mass partial: thread t<64 owns k=t ----
    if (tid < K_) {
      float s = 0.0f;
      #pragma unroll
      for (int j = 0; j < NT; ++j) s += scoresF[tid * NT + j];
      massReg += s;
    }

    // ---- agg GEMM: one 16x16x32 WMMA per accumulator tile ----
    #pragma unroll
    for (int i = 0; i < 4; ++i) {
      Frag fa;
      const __bf16* ar = xdn + ((wave * 4 + i) * 16 + lr) * NT;   // A = x[16 x 32]
      fa.q[0] = *(const float4*)(ar + 8 * hi);
      fa.q[1] = *(const float4*)(ar + 16 + 8 * hi);
      #pragma unroll
      for (int j = 0; j < 4; ++j) {
        Frag fb;
        const __bf16* br = sbf + (j * 16 + lr) * NT;              // B = s[32 x 16]
        fb.q[0] = *(const float4*)(br + 8 * hi);
        fb.q[1] = *(const float4*)(br + 16 + 8 * hi);
        acc[i][j] = __builtin_amdgcn_wmma_f32_16x16x32_bf16(false, fa.v, false, fb.v,
                                                            (short)0, acc[i][j],
                                                            false, false);
      }
    }
  }

  // ---- flush partials: f32 atomics (16 N-chunks reduce into agg/mass) ----
  if (tid < K_) atomicAdd(&mass[b * K_ + tid], massReg);
  #pragma unroll
  for (int i = 0; i < 4; ++i)
    #pragma unroll
    for (int j = 0; j < 4; ++j)
      #pragma unroll
      for (int e = 0; e < 8; ++e) {
        int d = (wave * 4 + i) * 16 + e + 8 * hi;
        int k = j * 16 + lr;
        atomicAdd(&agg[((size_t)b * D_ + d) * K_ + k], acc[i][j][e]);
      }
}

// ---------------------------------------------------------------------------
// Kernel 2: per-(b,k) column: desc = agg - centers*mass, intra-normalize,
//           write to out, accumulate global sum-of-squares per batch.
//   grid = (K, B), 128 threads
// ---------------------------------------------------------------------------
__global__ __launch_bounds__(128)
void netvlad_colnorm(const float* __restrict__ agg,      // [B,D,K]
                     const float* __restrict__ mass,     // [B,K]
                     const float* __restrict__ centers,  // [D,K]
                     float* __restrict__ out,            // [B, D*K]
                     float* __restrict__ gss)            // [B]
{
  __shared__ float red[128];
  const int k = blockIdx.x, b = blockIdx.y, tid = threadIdx.x;
  const float m = mass[b * K_ + k];
  float v[4]; float ss = 0.0f;
  #pragma unroll
  for (int i = 0; i < 4; ++i) {
    int d = tid + i * 128;
    float a = agg[((size_t)b * D_ + d) * K_ + k];
    float c = centers[d * K_ + k];
    v[i] = a - c * m;
    ss += v[i] * v[i];
  }
  red[tid] = ss; __syncthreads();
  for (int s = 64; s > 0; s >>= 1) {
    if (tid < s) red[tid] += red[tid + s];
    __syncthreads();
  }
  float sum = red[0];
  float inv = 1.0f / fmaxf(sqrtf(sum), 1e-12f);
  #pragma unroll
  for (int i = 0; i < 4; ++i) {
    int d = tid + i * 128;
    out[(size_t)b * (D_ * K_) + d * K_ + k] = v[i] * inv;  // flatten: d*K + k
  }
  if (tid == 0) atomicAdd(&gss[b], sum * inv * inv);       // ||normalized col||^2
}

// ---------------------------------------------------------------------------
// Kernel 3: global L2 normalize per batch (in place on out)
// ---------------------------------------------------------------------------
__global__ void netvlad_gnorm(float* __restrict__ out, const float* __restrict__ gss) {
  int i = blockIdx.x * blockDim.x + threadIdx.x;
  if (i < B_ * D_ * K_) {
    int b = i >> 15;   // D*K = 32768
    float inv = 1.0f / fmaxf(sqrtf(gss[b]), 1e-12f);
    out[i] *= inv;
  }
}

// ---------------------------------------------------------------------------
extern "C" void kernel_launch(void* const* d_in, const int* in_sizes, int n_in,
                              void* d_out, int out_size, void* d_ws, size_t ws_size,
                              hipStream_t stream) {
  (void)in_sizes; (void)n_in; (void)out_size; (void)ws_size;
  const float* x       = (const float*)d_in[0];   // [B,D,N] f32
  const float* W       = (const float*)d_in[1];   // [K,D]   f32
  const float* centers = (const float*)d_in[2];   // [D,K]   f32
  float* out = (float*)d_out;                     // [B, D*K] f32

  float* agg  = (float*)d_ws;                     // B*D*K
  float* mass = agg + (size_t)B_ * D_ * K_;       // B*K
  float* gss  = mass + B_ * K_;                   // B

  const int zn = B_ * D_ * K_ + B_ * K_ + B_;
  netvlad_zero<<<(zn + 255) / 256, 256, 0, stream>>>(agg, zn);

  netvlad_fused<<<dim3(NCHUNK, B_), 256, SH_BYTES, stream>>>(x, W, agg, mass);

  netvlad_colnorm<<<dim3(K_, B_), 128, 0, stream>>>(agg, mass, centers, out, gss);
  netvlad_gnorm<<<(B_ * D_ * K_ + 255) / 256, 256, 0, stream>>>(out, gss);
}
